// NewestAttentionMLP_76338748719848
// MI455X (gfx1250) — matrix operand order repaired and checked by reading, not compiled, not run on hardware
//
#include <hip/hip_runtime.h>
#include <hip/hip_bf16.h>

typedef __attribute__((ext_vector_type(16))) __bf16 v16bf;
typedef __attribute__((ext_vector_type(8)))  float  v8f;
typedef unsigned int uint32;
typedef __attribute__((address_space(3))) unsigned char lds_u8_t;

#define EPSF 1e-5f

union CF  { v8f   v; float  f[8]; };
union F16 { v16bf v; uint32 u[8]; };

__device__ __forceinline__ unsigned lds_off_of(void* p) {
  return (unsigned)(unsigned long long)(lds_u8_t*)p;
}

__device__ __forceinline__ uint32 pk2(float a, float b) {
  union { __bf16 h[2]; uint32 u; } t;
  t.h[0] = (__bf16)a; t.h[1] = (__bf16)b;
  return t.u;
}

// 4 WMMA steps consuming a staged 128-deep K chunk
__device__ __forceinline__ void mma4(const __bf16* Arow, const __bf16* Brow,
                                     CF& c, int hf) {
#pragma unroll
  for (int s = 0; s < 4; ++s) {
    int kk = s * 32;
    F16 a, bq;
#pragma unroll
    for (int v = 0; v < 8; ++v) {
      int ka = kk + 2 * v + ((v >= 4) ? 8 : 0) + 8 * hf;   // ISA 16-bit A 16x32 layout
      a.u[v]  = *(const uint32*)(Arow + ka);
      int kb = kk + 2 * v + ((v >= 4) ? 8 : 0) + 8 * hf;   // same k order as A
      bq.u[v] = *(const uint32*)(Brow + kb);
    }
    c.v = __builtin_amdgcn_wmma_f32_16x16x32_bf16(
              false, a.v, false, bq.v, (short)0, c.v, false, false);
  }
}

// ---------------- row-wise LayerNorm -> bf16 (only used for x, K=1024) ----------------
__global__ __launch_bounds__(256) void ln_rows_kernel(
    const float* __restrict__ x, const float* __restrict__ g,
    const float* __restrict__ b, __bf16* __restrict__ out,
    int M, int K, int Kpad)
{
  int wid  = (blockIdx.x * blockDim.x + threadIdx.x) >> 5;
  int lane = threadIdx.x & 31;
  if (wid >= M) return;
  const float* row = x + (size_t)wid * K;
  float s = 0.f, q = 0.f;
  for (int i = lane; i < K; i += 32) { float v = row[i]; s += v; q += v * v; }
#pragma unroll
  for (int off = 16; off > 0; off >>= 1) {
    s += __shfl_xor(s, off, 32);
    q += __shfl_xor(q, off, 32);
  }
  float inv = 1.f / (float)K;
  float mu  = s * inv;
  float var = q * inv - mu * mu;
  float r   = rsqrtf(var + EPSF);
  __bf16* orow = out + (size_t)wid * Kpad;
  for (int i = lane; i < K; i += 32)
    orow[i] = (__bf16)((row[i] - mu) * r * g[i] + b[i]);
  for (int i = K + lane; i < Kpad; i += 32)
    orow[i] = (__bf16)0.f;
}

// ---------------- generic WMMA GEMM, 512 threads = 16 waves, tile 32(M) x 128(N) -------
// fuse==0: A is pre-normalized bf16 [M, lda]; K staged in 128-chunks via async-to-LDS.
// fuse==1: A is f32 [M, 128]; LayerNorm(K=128) fused into staging.
__global__ __launch_bounds__(512) void gemm_wmma_kernel(
    const void*  __restrict__ Aptr, int lda,
    const float* __restrict__ lng, const float* __restrict__ lnb,
    const float* __restrict__ W, int ldw,
    const float* __restrict__ bias,
    float*       __restrict__ out,
    int Nfull, int K, int Kpad, int relu, int fuse)
{
  __shared__ __bf16 As[32 * 136];
  __shared__ __bf16 Bt[128 * 136];

  const int tid   = threadIdx.x;
  const int lane  = tid & 31;
  const int wave  = tid >> 5;
  const int hf    = lane >> 4;
  const int l16   = lane & 15;
  const int m0    = (wave >> 3) * 16;
  const int n0l   = (wave & 7) * 16;
  const int mBase = blockIdx.x * 32;
  const int nBase = blockIdx.y * 128;

  CF c;
  const int nIdx = n0l + l16;
  float bv = (nIdx < Nfull) ? bias[nBase + nIdx] : 0.f;
#pragma unroll
  for (int r = 0; r < 8; ++r) c.f[r] = bv;

  const __bf16* Arow = As + (m0  + l16) * 136;
  const __bf16* Brow = Bt + (n0l + l16) * 136;

  if (fuse) {
    const int m = tid >> 4, j = tid & 15;
    const float* Ar = (const float*)Aptr + (size_t)(mBase + m) * lda;
    float4 v0 = *(const float4*)(Ar + j * 4);
    float4 v1 = *(const float4*)(Ar + 64 + j * 4);
    float s = v0.x + v0.y + v0.z + v0.w + v1.x + v1.y + v1.z + v1.w;
    float q = v0.x*v0.x + v0.y*v0.y + v0.z*v0.z + v0.w*v0.w
            + v1.x*v1.x + v1.y*v1.y + v1.z*v1.z + v1.w*v1.w;
#pragma unroll
    for (int off = 1; off < 16; off <<= 1) {
      s += __shfl_xor(s, off, 32);
      q += __shfl_xor(q, off, 32);
    }
    float mu  = s * (1.f / 128.f);
    float var = q * (1.f / 128.f) - mu * mu;
    float rr  = rsqrtf(var + EPSF);
    int c0 = j * 4, c1 = 64 + j * 4;
    uint32* d0 = (uint32*)(As + m * 136 + c0);
    uint32* d1 = (uint32*)(As + m * 136 + c1);
    d0[0] = pk2((v0.x - mu) * rr * lng[c0]     + lnb[c0],
                (v0.y - mu) * rr * lng[c0 + 1] + lnb[c0 + 1]);
    d0[1] = pk2((v0.z - mu) * rr * lng[c0 + 2] + lnb[c0 + 2],
                (v0.w - mu) * rr * lng[c0 + 3] + lnb[c0 + 3]);
    d1[0] = pk2((v1.x - mu) * rr * lng[c1]     + lnb[c1],
                (v1.y - mu) * rr * lng[c1 + 1] + lnb[c1 + 1]);
    d1[1] = pk2((v1.z - mu) * rr * lng[c1 + 2] + lnb[c1 + 2],
                (v1.w - mu) * rr * lng[c1 + 3] + lnb[c1 + 3]);
    for (int t = tid; t < 4096; t += 512) {
      int k = t >> 5, n = (t & 31) * 4;
      if (n < Nfull) {
        float4 wv = *(const float4*)(W + (size_t)k * ldw + nBase + n);
        Bt[(n    ) * 136 + k] = (__bf16)wv.x;
        Bt[(n + 1) * 136 + k] = (__bf16)wv.y;
        Bt[(n + 2) * 136 + k] = (__bf16)wv.z;
        Bt[(n + 3) * 136 + k] = (__bf16)wv.w;
      }
    }
    __syncthreads();
    mma4(Arow, Brow, c, hf);
  } else {
    const int m = tid >> 4, cidx = tid & 15;
    unsigned long long gaBase =
        (unsigned long long)((const __bf16*)Aptr + (size_t)(mBase + m) * lda + cidx * 8);
    unsigned asA = lds_off_of((void*)As) + m * 272 + cidx * 16;
    for (int kc = 0; kc < Kpad; kc += 128) {
      unsigned long long ga = gaBase + (unsigned long long)kc * 2;
      asm volatile("global_load_async_to_lds_b128 %0, %1, off"
                   :: "v"(asA), "v"(ga) : "memory");
      for (int t = tid; t < 4096; t += 512) {
        int k = t >> 5, n = (t & 31) * 4;
        int kg = kc + k;
        if (n < Nfull) {
          if (kg < K) {
            float4 wv = *(const float4*)(W + (size_t)kg * ldw + nBase + n);
            Bt[(n    ) * 136 + k] = (__bf16)wv.x;
            Bt[(n + 1) * 136 + k] = (__bf16)wv.y;
            Bt[(n + 2) * 136 + k] = (__bf16)wv.z;
            Bt[(n + 3) * 136 + k] = (__bf16)wv.w;
          } else {
            Bt[(n    ) * 136 + k] = (__bf16)0.f;
            Bt[(n + 1) * 136 + k] = (__bf16)0.f;
            Bt[(n + 2) * 136 + k] = (__bf16)0.f;
            Bt[(n + 3) * 136 + k] = (__bf16)0.f;
          }
        }
      }
      asm volatile("s_wait_asynccnt 0x0" ::: "memory");
      __syncthreads();
      mma4(Arow, Brow, c, hf);
      __syncthreads();
    }
  }

  if (nIdx < Nfull) {
    const int n = nBase + nIdx;
#pragma unroll
    for (int r = 0; r < 8; ++r) {
      int ml = m0 + r + 8 * hf;
      float v = c.f[r];
      if (relu) v = fmaxf(v, 0.f);
      out[(size_t)(mBase + ml) * Nfull + n] = v;
    }
  }
}

// ------- fused residual stack: h kept resident in LDS across 4 blocks + fc proj -------
// per block: 32 full rows; it<4: Hs += relu(LN(Hs)@Wr[it]+br[it]); it==4: fc = LN(Hs)@Wout+bout
__global__ __launch_bounds__(512) void residual_stack_kernel(
    const float* __restrict__ hin,                                   // [M,128]
    const float* __restrict__ lnr_g, const float* __restrict__ lnr_b,// [4,128]
    const float* __restrict__ Wr,    const float* __restrict__ br,   // [4,128,128],[4,128]
    const float* __restrict__ lnf_g, const float* __restrict__ lnf_b,// [128]
    const float* __restrict__ Wout,  const float* __restrict__ bout, // [128,64],[64]
    float* __restrict__ fc)                                          // [M,64]
{
  __shared__ __bf16 As[32 * 136];
  __shared__ __bf16 Bt[128 * 136];
  __shared__ float  Hs[32 * 132];

  const int tid   = threadIdx.x;
  const int lane  = tid & 31;
  const int wave  = tid >> 5;
  const int hf    = lane >> 4;
  const int l16   = lane & 15;
  const int m0    = (wave >> 3) * 16;
  const int n0l   = (wave & 7) * 16;
  const int mBase = blockIdx.x * 32;
  const int nIdx  = n0l + l16;
  const int m = tid >> 4, j = tid & 15;

  const __bf16* Arow = As + (m0  + l16) * 136;
  const __bf16* Brow = Bt + (n0l + l16) * 136;

  // initial 32x128 h tile -> registers (and Hs)
  float4 v0, v1;
  {
    const float* Ar = hin + (size_t)(mBase + m) * 128;
    v0 = *(const float4*)(Ar + j * 4);
    v1 = *(const float4*)(Ar + 64 + j * 4);
    *(float4*)(Hs + m * 132 + j * 4)      = v0;
    *(float4*)(Hs + m * 132 + 64 + j * 4) = v1;
  }

  for (int it = 0; it < 5; ++it) {
    const int last = (it == 4);
    const float* g   = last ? lnf_g : lnr_g + it * 128;
    const float* be  = last ? lnf_b : lnr_b + it * 128;
    const float* Wp  = last ? Wout  : Wr + (size_t)it * 128 * 128;
    const float* bp  = last ? bout  : br + it * 128;
    const int Ncur   = last ? 64 : 128;
    const int ldw    = Ncur;

    if (it > 0) {   // refreshed tile (barrier at end of previous iteration)
      v0 = *(const float4*)(Hs + m * 132 + j * 4);
      v1 = *(const float4*)(Hs + m * 132 + 64 + j * 4);
    }
    // row stats over 16-lane row group
    float s = v0.x + v0.y + v0.z + v0.w + v1.x + v1.y + v1.z + v1.w;
    float q = v0.x*v0.x + v0.y*v0.y + v0.z*v0.z + v0.w*v0.w
            + v1.x*v1.x + v1.y*v1.y + v1.z*v1.z + v1.w*v1.w;
#pragma unroll
    for (int off = 1; off < 16; off <<= 1) {
      s += __shfl_xor(s, off, 32);
      q += __shfl_xor(q, off, 32);
    }
    float mu  = s * (1.f / 128.f);
    float var = q * (1.f / 128.f) - mu * mu;
    float rr  = rsqrtf(var + EPSF);
    int c0 = j * 4, c1 = 64 + j * 4;
    uint32* d0 = (uint32*)(As + m * 136 + c0);
    uint32* d1 = (uint32*)(As + m * 136 + c1);
    d0[0] = pk2((v0.x - mu) * rr * g[c0]     + be[c0],
                (v0.y - mu) * rr * g[c0 + 1] + be[c0 + 1]);
    d0[1] = pk2((v0.z - mu) * rr * g[c0 + 2] + be[c0 + 2],
                (v0.w - mu) * rr * g[c0 + 3] + be[c0 + 3]);
    d1[0] = pk2((v1.x - mu) * rr * g[c1]     + be[c1],
                (v1.y - mu) * rr * g[c1 + 1] + be[c1 + 1]);
    d1[1] = pk2((v1.z - mu) * rr * g[c1 + 2] + be[c1 + 2],
                (v1.w - mu) * rr * g[c1 + 3] + be[c1 + 3]);
    // stage weights
    for (int t = tid; t < 4096; t += 512) {
      int k = t >> 5, n = (t & 31) * 4;
      if (n < Ncur) {
        float4 wv = *(const float4*)(Wp + (size_t)k * ldw + n);
        Bt[(n    ) * 136 + k] = (__bf16)wv.x;
        Bt[(n + 1) * 136 + k] = (__bf16)wv.y;
        Bt[(n + 2) * 136 + k] = (__bf16)wv.z;
        Bt[(n + 3) * 136 + k] = (__bf16)wv.w;
      }
    }
    __syncthreads();

    CF c;
    float bv = (nIdx < Ncur) ? bp[nIdx] : 0.f;
#pragma unroll
    for (int r = 0; r < 8; ++r) c.f[r] = bv;
    mma4(Arow, Brow, c, hf);

    if (!last) {
      // per-lane exclusive read-modify-write of Hs elements
#pragma unroll
      for (int r = 0; r < 8; ++r) {
        int ml = m0 + r + 8 * hf;
        float old = Hs[ml * 132 + nIdx];
        Hs[ml * 132 + nIdx] = old + fmaxf(c.f[r], 0.f);
      }
      __syncthreads();   // Hs updates + As/Bt reads complete before next iteration
    } else if (nIdx < 64) {
#pragma unroll
      for (int r = 0; r < 8; ++r) {
        int ml = m0 + r + 8 * hf;
        fc[(size_t)(mBase + ml) * 64 + nIdx] = c.f[r];
      }
    }
  }
}

// ------- fused: t[b,o,:] = {fc[b,l,o], mean}, LN(513) -> bf16 padded to Kpad -------
__global__ __launch_bounds__(256) void build_t_ln_kernel(
    const float* __restrict__ fc,
    const float* __restrict__ g, const float* __restrict__ b,
    __bf16* __restrict__ out, int Kpad)
{
  int rid  = (blockIdx.x * blockDim.x + threadIdx.x) >> 5;
  int lane = threadIdx.x & 31;
  if (rid >= 256 * 64) return;
  int bb = rid >> 6, o = rid & 63;
  const float* base = fc + (size_t)bb * 512 * 64 + o;
  float v[16];
  float s = 0.f, q = 0.f;
#pragma unroll
  for (int i = 0; i < 16; ++i) {
    v[i] = base[(size_t)(lane + 32 * i) * 64];
    s += v[i]; q += v[i] * v[i];
  }
#pragma unroll
  for (int off = 16; off > 0; off >>= 1) {
    s += __shfl_xor(s, off, 32);
    q += __shfl_xor(q, off, 32);
  }
  float me  = s * (1.f / 512.f);
  float S   = s + me, Q = q + me * me;
  float mu  = S * (1.f / 513.f);
  float var = Q * (1.f / 513.f) - mu * mu;
  float r   = rsqrtf(var + EPSF);
  __bf16* orow = out + (size_t)rid * Kpad;
#pragma unroll
  for (int i = 0; i < 16; ++i) {
    int l = lane + 32 * i;
    orow[l] = (__bf16)((v[i] - mu) * r * g[l] + b[l]);
  }
  if (lane == 0)
    orow[512] = (__bf16)((me - mu) * r * g[512] + b[512]);
  for (int i = 513 + lane; i < Kpad; i += 32)
    orow[i] = (__bf16)0.f;
}

// ------- fused softmax(512) + weighted reduction -> out[b,o] -------
__global__ __launch_bounds__(256) void softmax_out_kernel(
    const float* __restrict__ w,
    const float* __restrict__ fc,
    float* __restrict__ out)
{
  int rid  = (blockIdx.x * blockDim.x + threadIdx.x) >> 5;
  int lane = threadIdx.x & 31;
  if (rid >= 256 * 64) return;
  int bb = rid >> 6, o = rid & 63;
  const float* wr = w + (size_t)rid * 512;
  float x[16];
  float mx = -3.402823466e38f;
#pragma unroll
  for (int i = 0; i < 16; ++i) { x[i] = wr[lane + 32 * i]; mx = fmaxf(mx, x[i]); }
#pragma unroll
  for (int off = 16; off > 0; off >>= 1) mx = fmaxf(mx, __shfl_xor(mx, off, 32));
  const float* base = fc + (size_t)bb * 512 * 64 + o;
  float s = 0.f, d = 0.f;
#pragma unroll
  for (int i = 0; i < 16; ++i) {
    float e = __expf(x[i] - mx);
    s += e;
    d += e * base[(size_t)(lane + 32 * i) * 64];
  }
#pragma unroll
  for (int off = 16; off > 0; off >>= 1) {
    s += __shfl_xor(s, off, 32);
    d += __shfl_xor(d, off, 32);
  }
  if (lane == 0) out[rid] = d / s;
}

extern "C" void kernel_launch(void* const* d_in, const int* in_sizes, int n_in,
                              void* d_out, int out_size, void* d_ws, size_t ws_size,
                              hipStream_t stream)
{
  (void)in_sizes; (void)n_in; (void)out_size; (void)ws_size;
  const float* x      = (const float*)d_in[0];
  const float* ln0_g  = (const float*)d_in[1];
  const float* ln0_b  = (const float*)d_in[2];
  const float* W1     = (const float*)d_in[3];
  const float* b1     = (const float*)d_in[4];
  const float* lnr_g  = (const float*)d_in[5];
  const float* lnr_b  = (const float*)d_in[6];
  const float* Wr     = (const float*)d_in[7];
  const float* br     = (const float*)d_in[8];
  const float* lnf_g  = (const float*)d_in[9];
  const float* lnf_b  = (const float*)d_in[10];
  const float* Wout   = (const float*)d_in[11];
  const float* bout   = (const float*)d_in[12];
  const float* aln0_g = (const float*)d_in[13];
  const float* aln0_b = (const float*)d_in[14];
  const float* Wa1    = (const float*)d_in[15];
  const float* ba1    = (const float*)d_in[16];
  const float* aln1_g = (const float*)d_in[17];
  const float* aln1_b = (const float*)d_in[18];
  const float* Wa2    = (const float*)d_in[19];
  const float* ba2    = (const float*)d_in[20];
  const float* aln2_g = (const float*)d_in[21];
  const float* aln2_b = (const float*)d_in[22];
  const float* Wa3    = (const float*)d_in[23];
  const float* ba3    = (const float*)d_in[24];

  const size_t M1 = 131072;  // BS*LN
  const size_t M2 = 16384;   // BS*OUT

  char*  ws   = (char*)d_ws;
  __bf16* xln = (__bf16*)(ws + 0);             // 131072 x 1024 bf16 = 256 MB
  float*  h   = (float*) (ws + 268435456ull);  // 131072 x 128 f32   =  64 MB
  // recycle dead xln region for stage 2/3:
  float*  fc  = (float*) (ws + 0);             // 131072 x 64 f32    =  32 MB
  __bf16* tln = (__bf16*)(ws + 33554432ull);   // 16384 x 640 bf16   =  20 MB
  float*  a1  = (float*) (ws + 54525952ull);   // 16384 x 128 f32
  float*  a2  = (float*) (ws + 62914560ull);   // 16384 x 128 f32
  float*  wsm = (float*) (ws + 71303168ull);   // 16384 x 512 f32    =  32 MB

  // ---- fc stack ----
  ln_rows_kernel<<<(unsigned)(M1 / 8), 256, 0, stream>>>(x, ln0_g, ln0_b, xln, (int)M1, 1024, 1024);
  gemm_wmma_kernel<<<dim3((unsigned)(M1 / 32), 1), 512, 0, stream>>>(
      xln, 1024, nullptr, nullptr, W1, 128, b1, h, 128, 1024, 1024, 1, 0);
  residual_stack_kernel<<<(unsigned)(M1 / 32), 512, 0, stream>>>(
      h, lnr_g, lnr_b, Wr, br, lnf_g, lnf_b, Wout, bout, fc);

  // ---- attention-weight MLP ----
  build_t_ln_kernel<<<(unsigned)(M2 / 8), 256, 0, stream>>>(fc, aln0_g, aln0_b, tln, 640);
  gemm_wmma_kernel<<<dim3((unsigned)(M2 / 32), 1), 512, 0, stream>>>(
      tln, 640, nullptr, nullptr, Wa1, 128, ba1, a1, 128, 513, 640, 1, 0);
  gemm_wmma_kernel<<<dim3((unsigned)(M2 / 32), 1), 512, 0, stream>>>(
      a1, 128, aln1_g, aln1_b, Wa2, 128, ba2, a2, 128, 128, 128, 1, 1);
  gemm_wmma_kernel<<<dim3((unsigned)(M2 / 32), 4), 512, 0, stream>>>(
      a2, 128, aln2_g, aln2_b, Wa3, 512, ba3, wsm, 512, 128, 128, 0, 1);

  // ---- softmax + weighted reduction ----
  softmax_out_kernel<<<(unsigned)(M2 / 8), 256, 0, stream>>>(wsm, fc, (float*)d_out);
}
